// AnomalyAwareMemory_11596411699522
// MI455X (gfx1250) — matrix operand order repaired and checked by reading, not compiled, not run on hardware
//
#include <hip/hip_runtime.h>
#include <math.h>
#include <stdint.h>

// Problem sizes (fixed by the reference)
#define N_    2048
#define D_    256
#define MEM_  16384

typedef __attribute__((ext_vector_type(16))) _Float16 v16h;
typedef __attribute__((ext_vector_type(8)))  _Float16 v8h;
typedef __attribute__((ext_vector_type(8)))  float    v8f;
typedef __attribute__((ext_vector_type(4)))  unsigned int v4u;
typedef __attribute__((ext_vector_type(8)))  int      v8i;
typedef __attribute__((ext_vector_type(4)))  int      v4i;

// Tensor Data Mover availability (arity differs across toolchains)
#if defined(__has_builtin)
#  if __has_builtin(__builtin_amdgcn_tensor_load_to_lds) && __has_builtin(__builtin_amdgcn_s_wait_tensorcnt)
#    define HAVE_TDM 1
#  endif
#endif

// ---------------- workspace layout (float units) ----------------
// total ws requirement ~= 36 MB
static constexpr size_t WS_MU    = 0;                       // 256
static constexpr size_t WS_RM    = 256;                     // 256
static constexpr size_t WS_G     = 512;                     // 65536 (Z^T Z)
static constexpr size_t WS_RC    = WS_G + 65536;            // 65536 (EMA cov + 1e-6 I)
static constexpr size_t WS_CINV  = WS_RC + 65536;           // 65536
static constexpr size_t WS_DIST  = WS_CINV + 65536;         // 2048
static constexpr size_t WS_STATS = WS_DIST + 2048;          // dmin, dmax, kl
static constexpr size_t WS_IMP   = WS_STATS + 64;           // 2048 sorted importance (desc)
static constexpr size_t WS_ORD   = WS_IMP + 2048;           // 2048 ints (argsort)
static constexpr size_t WS_MEMWS = 262144;                  // updated memory [MEM,D] f32
static constexpr size_t WS_HALF  = WS_MEMWS + (size_t)MEM_ * D_; // f16 region start
// half-unit offsets inside the f16 region
static constexpr size_t H_Q = 0;
static constexpr size_t H_K = (size_t)N_ * D_;
static constexpr size_t H_V = H_K + (size_t)MEM_ * D_;

// ---------------- WMMA fragment loaders (wave32, 16x16x32 f16) ----------------
// A (16x32, M=lane%16): per lane, K = hi*16+e  -> 16 contiguous halves
// B (32x16, N=lane%16): per lane, K = hi*16+e  -> 16 contiguous halves
__device__ __forceinline__ v16h cat8(v8h lo, v8h hi8) {
  return __builtin_shufflevector(lo, hi8, 0,1,2,3,4,5,6,7,8,9,10,11,12,13,14,15);
}
__device__ __forceinline__ v16h afrag_h(const _Float16* p, int hi) {
  return *(const v16h*)(p + hi * 16);
}
__device__ __forceinline__ v16h bfrag_h(const _Float16* p, int hi) {
  return *(const v16h*)(p + hi * 16);
}
__device__ __forceinline__ v16h afrag_f32(const float* p, int hi) {
  v16h r;
#pragma unroll
  for (int e = 0; e < 16; ++e) r[e] = (_Float16)p[hi * 16 + e];
  return r;
}
__device__ __forceinline__ v16h bfrag_f32(const float* p, int hi) {
  v16h r;
#pragma unroll
  for (int e = 0; e < 16; ++e) r[e] = (_Float16)p[hi * 16 + e];
  return r;
}

#ifdef HAVE_TDM
// 1-D contiguous DMA: global -> LDS, nbytes must be a multiple of 8.
// D# per ISA ch.8: group0 = {count=1, lds_addr, global_addr, type=2},
// group1 = {data_size=8B, tensor_dim0=tile_dim0=nbytes/8, stride=nbytes/8}.
__device__ __forceinline__ void tdm_load_1d(unsigned lds_addr, const void* gptr,
                                            unsigned nbytes) {
  unsigned long long ga = (unsigned long long)(uintptr_t)gptr;
  unsigned n8 = nbytes >> 3;
  v4u g0;
  g0[0] = 1u;                                                   // count=1, user D#
  g0[1] = lds_addr;                                             // LDS byte address
  g0[2] = (unsigned)ga;                                         // global_addr[31:0]
  g0[3] = (unsigned)((ga >> 32) & 0x01FFFFFFu) | (2u << 30);    // addr[56:32] | type=2
  v8i g1;
  g1[0] = (int)(3u << 16);          // workgroup_mask=0, data_size=3 (8 bytes)
  g1[1] = (int)((n8 & 0xFFFFu) << 16);   // tensor_dim0[15:0] in bits[79:64... (w1[31:16])
  g1[2] = (int)((n8 >> 16) & 0xFFFFu);   // tensor_dim0 high bits
  g1[3] = (int)((n8 & 0xFFFFu) << 16);   // tile_dim0 (16-bit) in w3[31:16]
  g1[4] = 0;                              // tile_dim1 = tile_dim2 = 0 (unused)
  g1[5] = (int)n8;                        // tensor_dim0_stride low 32
  g1[6] = 0;
  g1[7] = 0;
  v4i zz = {0, 0, 0, 0};
#if defined(__clang_major__) && (__clang_major__ >= 23)
  v8i z8 = {0, 0, 0, 0, 0, 0, 0, 0};
  __builtin_amdgcn_tensor_load_to_lds(g0, g1, zz, zz, z8, 0);
#else
  __builtin_amdgcn_tensor_load_to_lds(g0, g1, zz, zz, 0);
#endif
}
#endif

// ---------------- 1: column mean + EMA mean ----------------
__global__ void k_mean(const float* __restrict__ z, const float* __restrict__ rmean,
                       float* __restrict__ ws) {
  int d = threadIdx.x;
  float s = 0.f;
  for (int n = 0; n < N_; ++n) s += z[(size_t)n * D_ + d];
  float mu = s * (1.0f / N_);
  ws[WS_MU + d] = mu;
  ws[WS_RM + d] = 0.99f * rmean[d] + 0.01f * mu;   // MOMENTUM = 0.01
}

// ---------------- 2: G = Z^T Z (LDS tiled) ----------------
__global__ void k_gram(const float* __restrict__ z, float* __restrict__ ws) {
  __shared__ float si[32][16];
  __shared__ float sj[32][16];
  int tx = threadIdx.x, ty = threadIdx.y;
  int bi = blockIdx.y, bj = blockIdx.x;
  int t = ty * 16 + tx;
  float acc = 0.f;
  for (int kb = 0; kb < N_ / 32; ++kb) {
    for (int u = t; u < 512; u += 256) {
      int rr = u >> 4, cc = u & 15;
      si[rr][cc] = z[(size_t)(kb * 32 + rr) * D_ + bi * 16 + cc];
      sj[rr][cc] = z[(size_t)(kb * 32 + rr) * D_ + bj * 16 + cc];
    }
    __syncthreads();
#pragma unroll
    for (int kk = 0; kk < 32; ++kk) acc = fmaf(si[kk][ty], sj[kk][tx], acc);
    __syncthreads();
  }
  ws[WS_G + (size_t)(bi * 16 + ty) * D_ + bj * 16 + tx] = acc;
}

// ---------------- 3: rc = 0.99*RC + 0.01*cov + 1e-6 I ----------------
__global__ void k_rc(const float* __restrict__ rcov, float* __restrict__ ws) {
  int i = blockIdx.x, j = threadIdx.x;
  float cov = (ws[WS_G + (size_t)i * D_ + j] - (float)N_ * ws[WS_MU + i] * ws[WS_MU + j])
              * (1.0f / (float)(N_ - 1));
  float v = 0.99f * rcov[(size_t)i * D_ + j] + 0.01f * cov;
  if (i == j) v += 1e-6f;
  ws[WS_RC + (size_t)i * D_ + j] = v;
}

// ---------------- 4: Gauss-Jordan inverse in 256KB dynamic LDS ----------------
__global__ void __launch_bounds__(1024) k_invert(float* __restrict__ ws) {
  extern __shared__ float sA[];            // 256*256 floats
  __shared__ float fcol[256];
  __shared__ float s_ip;
  int tid = threadIdx.x;
  for (int u = tid; u < 65536; u += 1024) sA[u] = ws[WS_RC + u];
  __syncthreads();
  for (int k = 0; k < 256; ++k) {
    if (tid < 256) fcol[tid] = sA[tid * 256 + k];
    if (tid == 0) s_ip = 1.0f / sA[k * 256 + k];
    __syncthreads();
    float ip = s_ip;
    if (tid < 256) {
      float pv = sA[k * 256 + tid];
      sA[k * 256 + tid] = (tid == k) ? ip : pv * ip;
    }
    __syncthreads();
    for (int u = tid; u < 65536; u += 1024) {
      int i = u >> 8, j = u & 255;
      if (i != k) {
        float f = fcol[i];
        sA[u] = (j == k) ? (-f * ip) : fmaf(-f, sA[k * 256 + j], sA[u]);
      }
    }
    __syncthreads();
  }
  for (int u = tid; u < 65536; u += 1024) ws[WS_CINV + u] = sA[u];
}

// ---------------- 5: Mahalanobis distance ----------------
__global__ void k_maha(const float* __restrict__ z, float* __restrict__ ws) {
  __shared__ float c[256];
  __shared__ float red[256];
  int n = blockIdx.x, j = threadIdx.x;
  c[j] = z[(size_t)n * D_ + j] - ws[WS_RM + j];
  __syncthreads();
  float t = 0.f;
  for (int i = 0; i < 256; ++i) t = fmaf(c[i], ws[WS_CINV + (size_t)i * D_ + j], t);
  red[j] = t * c[j];
  __syncthreads();
  for (int s = 128; s > 0; s >>= 1) {
    if (j < s) red[j] += red[j + s];
    __syncthreads();
  }
  if (j == 0) ws[WS_DIST + n] = sqrtf(fmaxf(red[0], 1e-8f));
}

// ---------------- 6: dist min/max + label KL ----------------
__global__ void __launch_bounds__(1024) k_stats(const int* __restrict__ labels,
                                                float* __restrict__ ws) {
  __shared__ float smn[32], smx[32];
  __shared__ int cnt[2];
  int tid = threadIdx.x;
  if (tid < 2) cnt[tid] = 0;
  __syncthreads();
  float a = ws[WS_DIST + tid], b = ws[WS_DIST + tid + 1024];
  float mn = fminf(a, b), mx = fmaxf(a, b);
#pragma unroll
  for (int m = 1; m < 32; m <<= 1) {
    mn = fminf(mn, __shfl_xor(mn, m, 32));
    mx = fmaxf(mx, __shfl_xor(mx, m, 32));
  }
  if ((tid & 31) == 0) { smn[tid >> 5] = mn; smx[tid >> 5] = mx; }
  atomicAdd(&cnt[labels[tid]], 1);
  atomicAdd(&cnt[labels[tid + 1024]], 1);
  __syncthreads();
  if (tid == 0) {
    float dmn = smn[0], dmx = smx[0];
    for (int q = 1; q < 32; ++q) { dmn = fminf(dmn, smn[q]); dmx = fmaxf(dmx, smx[q]); }
    float p0 = (float)cnt[0] / (float)N_, p1 = (float)cnt[1] / (float)N_;
    float kl = p0 * logf(fmaxf(p0 * 2.f, 1e-8f)) + p1 * logf(fmaxf(p1 * 2.f, 1e-8f));
    ws[WS_STATS + 0] = dmn;
    ws[WS_STATS + 1] = dmx;
    ws[WS_STATS + 2] = fmaxf(kl, 0.f);
  }
}

// ---------------- 7: bitonic argsort of importance (desc, stable) ----------------
__global__ void __launch_bounds__(1024) k_sort(float* __restrict__ ws) {
  __shared__ unsigned long long key[2048];
  int tid = threadIdx.x;
  float dmn = ws[WS_STATS + 0], dmx = ws[WS_STATS + 1], kl = ws[WS_STATS + 2];
  float inv = 1.f / (dmx - dmn + 1e-8f);
  for (int i = tid; i < 2048; i += 1024) {
    float an = (ws[WS_DIST + i] - dmn) * inv;
    float imp = (1.f + 1.0f * an) * kl;                 // ANOMALY_WEIGHT = 1
    unsigned int fb = __float_as_uint(imp);             // imp >= 0 -> order-preserving bits
    key[i] = ((unsigned long long)(~fb) << 32) | (unsigned int)i;  // asc u64 == desc imp, idx tiebreak
  }
  for (int k = 2; k <= 2048; k <<= 1)
    for (int j = k >> 1; j > 0; j >>= 1) {
      __syncthreads();
      for (int i = tid; i < 2048; i += 1024) {
        int p = i ^ j;
        if (p > i) {
          unsigned long long x = key[i], y = key[p];
          bool up = ((i & k) == 0);
          if (up ? (x > y) : (x < y)) { key[i] = y; key[p] = x; }
        }
      }
    }
  __syncthreads();
  int* ord = (int*)&ws[WS_ORD];
  for (int i = tid; i < 2048; i += 1024) {
    unsigned long long v = key[i];
    ws[WS_IMP + i] = __uint_as_float(~(unsigned int)(v >> 32));
    ord[i] = (int)(unsigned int)(v & 0xffffffffu);
  }
}

// ---------------- 8: copy memory -> workspace ----------------
__global__ void k_copymem(const float4* __restrict__ src, float* __restrict__ ws) {
  float4* dst = (float4*)&ws[WS_MEMWS];
  int i = blockIdx.x * blockDim.x + threadIdx.x;
  dst[i] = src[i];
}

// ---------------- 9: serial priority-queue eviction (early-exit exact) ----------------
__global__ void __launch_bounds__(1024) k_evict(const float* __restrict__ z,
                                                const float* __restrict__ mweights,
                                                float* __restrict__ ws) {
  extern __shared__ float w[];                 // 16384 floats (64 KB)
  __shared__ float pm[32];
  __shared__ int pj[32];
  __shared__ int s_done, s_j, s_src, s_rep;
  int tid = threadIdx.x, lane = tid & 31, wid = tid >> 5;
  for (int u = tid; u < MEM_; u += 1024) w[u] = mweights[u];
  if (tid == 0) s_done = 0;
  const int* ord = (const int*)&ws[WS_ORD];
  float* memws = &ws[WS_MEMWS];
  for (int i = 0; i < N_; ++i) {
    __syncthreads();
    if (s_done) break;                         // once a replace fails, all later fail (imp desc)
    float bv = 3.4e38f;
    int bj = MEM_;
    int base = tid * 16;
    for (int u = 0; u < 16; ++u) {
      float v = w[base + u];
      if (v < bv) { bv = v; bj = base + u; }   // strict < keeps first occurrence
    }
#pragma unroll
    for (int m = 1; m < 32; m <<= 1) {
      float ov = __shfl_xor(bv, m, 32);
      int oj = __shfl_xor(bj, m, 32);
      if (ov < bv || (ov == bv && oj < bj)) { bv = ov; bj = oj; }
    }
    if (lane == 0) { pm[wid] = bv; pj[wid] = bj; }
    __syncthreads();
    if (tid == 0) {
      float fv = pm[0]; int fj = pj[0];
      for (int q = 1; q < 32; ++q)
        if (pm[q] < fv || (pm[q] == fv && pj[q] < fj)) { fv = pm[q]; fj = pj[q]; }
      float imp = ws[WS_IMP + i];
      if (imp > fv) { w[fj] = imp; s_j = fj; s_src = ord[i]; s_rep = 1; }
      else { s_done = 1; s_rep = 0; }
    }
    __syncthreads();
    if (s_rep && tid < D_)
      memws[(size_t)s_j * D_ + tid] = z[(size_t)s_src * D_ + tid];
  }
}

// ---------------- 10: WMMA projection  Out_f16 = A @ W^T + b ----------------
__global__ void __launch_bounds__(32) k_proj(const float* __restrict__ A,
                                             const float* __restrict__ W,
                                             const float* __restrict__ bias,
                                             _Float16* __restrict__ Out) {
  int lane = threadIdx.x & 31;
  int l16 = lane & 15, hi = lane >> 4;
  int mt = blockIdx.x, nt = blockIdx.y;
  const float* arow = A + (size_t)(mt * 16 + l16) * D_;
  const float* wrow = W + (size_t)(nt * 16 + l16) * D_;
  v8f acc = {};
#pragma unroll
  for (int c = 0; c < 8; ++c) {
    v16h a = afrag_f32(arow + c * 32, hi);
    v16h b = bfrag_f32(wrow + c * 32, hi);
    acc = __builtin_amdgcn_wmma_f32_16x16x32_f16(false, a, false, b, (short)0, acc, false, false);
  }
  float bs = bias[nt * 16 + l16];
#pragma unroll
  for (int r = 0; r < 8; ++r) {
    int m = mt * 16 + r + 8 * hi;
    Out[(size_t)m * D_ + nt * 16 + l16] = (_Float16)(acc[r] + bs);
  }
}

// ---------------- 11: flash attention, out = z + 0.5 * softmax(QK^T*0.625) V ----------------
// dynamic LDS: 2x K buf (16KB each) + 2x V buf (16KB each) + P (8KB) = 73728 B
static constexpr size_t LDS_KBUF = 0;            // halves; buf b at b*8192
static constexpr size_t LDS_VBUF = 16384;        // halves; buf b at 16384 + b*8192
static constexpr size_t LDS_P    = 32768;        // halves
static constexpr size_t FLASH_LDS_BYTES = (32768 + 4096) * 2;

__global__ void __launch_bounds__(256) k_flash(const float* __restrict__ z,
                                               float* __restrict__ out,
                                               const float* __restrict__ ws) {
  extern __shared__ __align__(16) _Float16 smem[];
  const _Float16* hb = (const _Float16*)&ws[WS_HALF];
  const _Float16* Qh = hb + H_Q;
  const _Float16* Kh = hb + H_K;
  const _Float16* Vh = hb + H_V;
  int tid = threadIdx.x, lane = tid & 31, wv = tid >> 5;
  int l16 = lane & 15, hi = lane >> 4;
  int qbase = (blockIdx.x * 8 + wv) * 16;
  const float sc = 0.625f;            // 1 / (sqrt(256) * TEMP)
  const int NCH = MEM_ / 32;

  v16h qf[8];
#pragma unroll
  for (int c = 0; c < 8; ++c)
    qf[c] = afrag_h(Qh + (size_t)(qbase + l16) * D_ + c * 32, hi);

  v8f vzero = {};
  v8f accd[16];
#pragma unroll
  for (int d = 0; d < 16; ++d) accd[d] = vzero;
  float mrow[8], lrow[8];
#pragma unroll
  for (int r = 0; r < 8; ++r) { mrow[r] = -1e30f; lrow[r] = 0.f; }

  for (int kb = 0; kb < NCH; ++kb) {
    int cur = kb & 1;
#ifdef HAVE_TDM
    if (kb == 0) {      // prologue: DMA chunk 0 into buffer 0 and wait
      if (wv == 0) {
        tdm_load_1d((unsigned)(uintptr_t)(void*)(smem + LDS_KBUF), Kh, 32 * D_ * 2);
        tdm_load_1d((unsigned)(uintptr_t)(void*)(smem + LDS_VBUF), Vh, 32 * D_ * 2);
        __builtin_amdgcn_s_wait_tensorcnt((short)0);
      }
      __syncthreads();
    }
    // overlap: kick off DMA of chunk kb+1 into the other buffer while computing on kb
    if (wv == 0 && kb + 1 < NCH) {
      tdm_load_1d((unsigned)(uintptr_t)(void*)(smem + LDS_KBUF + (cur ^ 1) * 8192),
                  Kh + (size_t)(kb + 1) * 32 * D_, 32 * D_ * 2);
      tdm_load_1d((unsigned)(uintptr_t)(void*)(smem + LDS_VBUF + (cur ^ 1) * 8192),
                  Vh + (size_t)(kb + 1) * 32 * D_, 32 * D_ * 2);
    }
#else
    {   // synchronous staging fallback (b128 copies)
      const uint4* gk = (const uint4*)(Kh + (size_t)kb * 32 * D_);
      const uint4* gv = (const uint4*)(Vh + (size_t)kb * 32 * D_);
      uint4* lk = (uint4*)(smem + LDS_KBUF + cur * 8192);
      uint4* lv = (uint4*)(smem + LDS_VBUF + cur * 8192);
      for (int u = tid; u < 1024; u += 256) { lk[u] = gk[u]; lv[u] = gv[u]; }
      if (kb + 1 < NCH)
        __builtin_prefetch(Kh + (size_t)(kb + 1) * 32 * D_, 0, 1);
    }
    __syncthreads();
#endif
    const _Float16* sKc = smem + LDS_KBUF + cur * 8192;
    const _Float16* sVc = smem + LDS_VBUF + cur * 8192;
    _Float16* sP = smem + LDS_P;

    // S = Q (16x256) @ Kchunk^T (32x256) -> two 16x16 tiles
    v8f s0 = vzero, s1 = vzero;
#pragma unroll
    for (int c = 0; c < 8; ++c) {
      v16h b0 = bfrag_h(sKc + (size_t)l16 * D_ + c * 32, hi);
      v16h b1 = bfrag_h(sKc + (size_t)(16 + l16) * D_ + c * 32, hi);
      s0 = __builtin_amdgcn_wmma_f32_16x16x32_f16(false, qf[c], false, b0, (short)0, s0, false, false);
      s1 = __builtin_amdgcn_wmma_f32_16x16x32_f16(false, qf[c], false, b1, (short)0, s1, false, false);
    }

    // online softmax update (row stats live in (vgpr r, lane-half) slots)
    float f_[8];
#pragma unroll
    for (int r = 0; r < 8; ++r) {
      float a0 = s0[r] * sc, a1 = s1[r] * sc;
      float rmax = fmaxf(a0, a1);
#pragma unroll
      for (int m = 1; m < 16; m <<= 1) rmax = fmaxf(rmax, __shfl_xor(rmax, m, 32));
      float mnew = fmaxf(mrow[r], rmax);
      float f = __expf(mrow[r] - mnew);
      float p0 = __expf(a0 - mnew), p1 = __expf(a1 - mnew);
      float rsum = p0 + p1;
#pragma unroll
      for (int m = 1; m < 16; m <<= 1) rsum += __shfl_xor(rsum, m, 32);
      lrow[r] = lrow[r] * f + rsum;
      mrow[r] = mnew;
      f_[r] = f;
      int row = r + 8 * hi;
      sP[(wv * 16 + row) * 32 + l16]      = (_Float16)p0;
      sP[(wv * 16 + row) * 32 + 16 + l16] = (_Float16)p1;
    }
#pragma unroll
    for (int d = 0; d < 16; ++d)
#pragma unroll
      for (int r = 0; r < 8; ++r) accd[d][r] = accd[d][r] * f_[r];

    asm volatile("s_wait_dscnt 0" ::: "memory");   // same-wave LDS RAW (C-layout -> A-frag)

    v16h pa = afrag_h(sP + (size_t)(wv * 16 + l16) * 32, hi);
#pragma unroll
    for (int d = 0; d < 16; ++d) {
      v16h b;
#pragma unroll
      for (int e = 0; e < 16; ++e)
        b[e] = sVc[(size_t)(hi * 16 + e) * D_ + d * 16 + l16];
      accd[d] = __builtin_amdgcn_wmma_f32_16x16x32_f16(false, pa, false, b, (short)0, accd[d], false, false);
    }
#ifdef HAVE_TDM
    if (wv == 0 && kb + 1 < NCH)
      __builtin_amdgcn_s_wait_tensorcnt((short)0);   // next chunk landed
#endif
    __syncthreads();
  }

#pragma unroll
  for (int d = 0; d < 16; ++d)
#pragma unroll
    for (int r = 0; r < 8; ++r) {
      int m = qbase + r + 8 * hi;
      int col = d * 16 + l16;
      out[(size_t)m * D_ + col] = z[(size_t)m * D_ + col] + 0.5f * accd[d][r] / lrow[r];
    }
}

// ---------------- host: launch pipeline ----------------
extern "C" void kernel_launch(void* const* d_in, const int* in_sizes, int n_in,
                              void* d_out, int out_size, void* d_ws, size_t ws_size,
                              hipStream_t stream) {
  (void)in_sizes; (void)n_in; (void)out_size; (void)ws_size;
  const float* z        = (const float*)d_in[0];
  const int*   labels   = (const int*)d_in[1];
  const float* memory   = (const float*)d_in[2];
  const float* mweights = (const float*)d_in[3];
  const float* rmean    = (const float*)d_in[4];
  const float* rcov     = (const float*)d_in[5];
  const float* Wq = (const float*)d_in[6];
  const float* bq = (const float*)d_in[7];
  const float* Wk = (const float*)d_in[8];
  const float* bk = (const float*)d_in[9];
  const float* Wv = (const float*)d_in[10];
  const float* bv = (const float*)d_in[11];
  float* out = (float*)d_out;
  float* ws  = (float*)d_ws;
  _Float16* hb = (_Float16*)&ws[WS_HALF];

  k_mean   <<<dim3(1),        dim3(256),     0,               stream>>>(z, rmean, ws);
  k_gram   <<<dim3(16, 16),   dim3(16, 16),  0,               stream>>>(z, ws);
  k_rc     <<<dim3(256),      dim3(256),     0,               stream>>>(rcov, ws);
  k_invert <<<dim3(1),        dim3(1024),    256*256*4,       stream>>>(ws);
  k_maha   <<<dim3(2048),     dim3(256),     0,               stream>>>(z, ws);
  k_stats  <<<dim3(1),        dim3(1024),    0,               stream>>>(labels, ws);
  k_sort   <<<dim3(1),        dim3(1024),    0,               stream>>>(ws);
  k_copymem<<<dim3(4096),     dim3(256),     0,               stream>>>((const float4*)memory, ws);
  k_evict  <<<dim3(1),        dim3(1024),    MEM_*4,          stream>>>(z, mweights, ws);
  k_proj   <<<dim3(128, 16),  dim3(32),      0,               stream>>>(z,             Wq, bq, hb + H_Q);
  k_proj   <<<dim3(1024, 16), dim3(32),      0,               stream>>>(&ws[WS_MEMWS], Wk, bk, hb + H_K);
  k_proj   <<<dim3(1024, 16), dim3(32),      0,               stream>>>(&ws[WS_MEMWS], Wv, bv, hb + H_V);
  k_flash  <<<dim3(16),       dim3(256),     FLASH_LDS_BYTES, stream>>>(z, out, ws);
}
